// AttentionWithBias_3504693314247
// MI455X (gfx1250) — matrix operand order repaired and checked by reading, not compiled, not run on hardware
//
#include <hip/hip_runtime.h>
#include <hip/hip_bf16.h>

// Problem dims (fixed by reference)
#define BB   4
#define LL   1024
#define II   32
#define DD   128
#define HH   8
#define DHH  32
#define HD   256    // HH*DHH

typedef __bf16 bf16_t;
typedef __attribute__((ext_vector_type(16))) __bf16 v16bf;
typedef __attribute__((ext_vector_type(8)))  __bf16 v8bf;
typedef __attribute__((ext_vector_type(8)))  float  v8f;

__device__ __forceinline__ v16bf concat8(v8bf a, v8bf b) {
  return __builtin_shufflevector(a, b, 0,1,2,3,4,5,6,7,8,9,10,11,12,13,14,15);
}
// 16 contiguous bf16 (B operand: one row of W / vT, contiguous K)
__device__ __forceinline__ v16bf ld16(const bf16_t* p) {
  return concat8(*(const v8bf*)p, *(const v8bf*)(p + 8));
}
// 16 contiguous bf16 (A operand: contiguous K {half*16 .. +15}; p1 unused)
__device__ __forceinline__ v16bf ld8x2(const bf16_t* p0, const bf16_t* p1) {
  return concat8(*(const v8bf*)p0, *(const v8bf*)(p0 + 8));
}
__device__ __forceinline__ v8f wmma_bf(v16bf a, v16bf b, v8f c) {
  return __builtin_amdgcn_wmma_f32_16x16x32_bf16(false, a, false, b, (short)0, c, false, false);
}

// ---------------- kernel 1: convert weights f32 -> bf16 ----------------
__global__ __launch_bounds__(256) void k_wcvt(
    const float* __restrict__ wqk, const float* __restrict__ wv,
    const float* __restrict__ wg,  const float* __restrict__ wo,
    bf16_t* __restrict__ oqk, bf16_t* __restrict__ ov,
    bf16_t* __restrict__ og,  bf16_t* __restrict__ oo) {
  int i = blockIdx.x * blockDim.x + threadIdx.x;   // 32768 threads
  if (i < HD * DD) {
    oqk[i] = (bf16_t)wqk[i];
    ov[i]  = (bf16_t)wv[i];
    og[i]  = (bf16_t)wg[i];
    oo[i]  = (bf16_t)wo[i];
  }
}

// ---------------- kernel 2: LN(x) + Q/K, V (and V^T), gate projections ----------------
// one wave per 16-token tile; 3 x (16x128 @ 128x256) via WMMA bf16
__global__ __launch_bounds__(32) void k_tok(
    const float* __restrict__ x, const float* __restrict__ g_in, const float* __restrict__ b_in,
    const bf16_t* __restrict__ wqk, const bf16_t* __restrict__ wv, const bf16_t* __restrict__ wg,
    const float* __restrict__ bg,
    bf16_t* __restrict__ qk, bf16_t* __restrict__ val, bf16_t* __restrict__ vT,
    float* __restrict__ gate) {
  __shared__ bf16_t xn[16 * 144];   // 16 tokens x 128 dims, padded stride 144
  const int lane = threadIdx.x, row = lane & 15, half = lane >> 4;
  const int b = blockIdx.x >> 6, q0 = (blockIdx.x & 63) << 4;
  const int tok = q0 + row;

  // LayerNorm over 128: two lanes per token (64 elems each)
  const float* xr = x + ((size_t)(b * LL + tok)) * DD + half * 64;
  float xv[64];
  float s = 0.f, s2 = 0.f;
#pragma unroll
  for (int i = 0; i < 64; ++i) { float v = xr[i]; xv[i] = v; s += v; s2 += v * v; }
  s  += __shfl_xor(s, 16);
  s2 += __shfl_xor(s2, 16);
  const float mean = s * (1.f / DD);
  const float rstd = rsqrtf(s2 * (1.f / DD) - mean * mean + 1e-5f);
#pragma unroll
  for (int i = 0; i < 64; ++i) {
    int d = half * 64 + i;
    xn[row * 144 + d] = (bf16_t)((xv[i] - mean) * rstd * g_in[d] + b_in[d]);
  }
  __syncthreads();

  // A operands (16 tokens x 32 dims per K-step)
  v16bf A[4];
#pragma unroll
  for (int ks = 0; ks < 4; ++ks) {
    const bf16_t* p = &xn[row * 144 + ks * 32 + half * 16];
    A[ks] = ld8x2(p, p + 16);
  }

  const bf16_t* Ws[3] = { wqk, wv, wg };
#pragma unroll
  for (int m = 0; m < 3; ++m) {
    const bf16_t* W = Ws[m];
    for (int nt = 0; nt < 16; ++nt) {
      v8f acc = {};
#pragma unroll
      for (int ks = 0; ks < 4; ++ks) {
        v16bf Bop = ld16(W + (size_t)(nt * 16 + row) * DD + ks * 32 + half * 16);
        acc = wmma_bf(A[ks], Bop, acc);
      }
      const int f = nt * 16 + row;
#pragma unroll
      for (int j = 0; j < 8; ++j) {
        const int t2 = q0 + j + 8 * half;
        const size_t ridx = ((size_t)(b * LL + t2)) * HD + f;
        if (m == 0) {
          qk[ridx] = (bf16_t)acc[j];
        } else if (m == 1) {
          val[ridx] = (bf16_t)acc[j];
          vT[((size_t)(b * HD + f)) * LL + t2] = (bf16_t)acc[j];
        } else {
          float gv = acc[j] + bg[f];
          gate[ridx] = 1.f / (1.f + __expf(-gv));
        }
      }
    }
  }
}

// ---------------- kernel 3: LN(bias) + 8-head projection (memory bound) ----------------
__global__ __launch_bounds__(256) void k_bias(
    const float* __restrict__ bias, const float* __restrict__ g_b, const float* __restrict__ b_b,
    const float* __restrict__ Wb, float* __restrict__ bproj) {
  const int gtid = blockIdx.x * blockDim.x + threadIdx.x;
  const int wid = gtid >> 5, lane = gtid & 31;
  const int nwaves = (gridDim.x * blockDim.x) >> 5;
  const int nrows = BB * LL * II;
  for (int r = wid; r < nrows; r += nwaves) {
    const float4 v = ((const float4*)(bias + (size_t)r * DD))[lane];
    float s = v.x + v.y + v.z + v.w;
    float s2 = v.x * v.x + v.y * v.y + v.z * v.z + v.w * v.w;
#pragma unroll
    for (int o = 16; o >= 1; o >>= 1) { s += __shfl_xor(s, o); s2 += __shfl_xor(s2, o); }
    const float mean = s * (1.f / DD);
    const float rstd = rsqrtf(s2 * (1.f / DD) - mean * mean + 1e-5f);
    const int d0 = lane * 4;
    float bn[4];
    bn[0] = (v.x - mean) * rstd * g_b[d0 + 0] + b_b[d0 + 0];
    bn[1] = (v.y - mean) * rstd * g_b[d0 + 1] + b_b[d0 + 1];
    bn[2] = (v.z - mean) * rstd * g_b[d0 + 2] + b_b[d0 + 2];
    bn[3] = (v.w - mean) * rstd * g_b[d0 + 3] + b_b[d0 + 3];
#pragma unroll
    for (int h = 0; h < HH; ++h) {
      const float4 w = ((const float4*)(Wb + h * DD))[lane];
      float p = bn[0] * w.x + bn[1] * w.y + bn[2] * w.z + bn[3] * w.w;
#pragma unroll
      for (int o = 16; o >= 1; o >>= 1) p += __shfl_xor(p, o);
      if (lane == 0) bproj[(size_t)r * HH + h] = p;
    }
  }
}

// ---------------- kernel 4: attention (unbiased), softmax over HEADS ----------------
// block = 8 waves, each wave: one 16-query tile x 128 keys; deterministic LDS reduce
__global__ __launch_bounds__(256) void k_attn(
    const bf16_t* __restrict__ qk, const bf16_t* __restrict__ vT,
    float* __restrict__ out_pre) {
  __shared__ bf16_t p_lds[8][16 * 40];   // per-wave P transpose staging
  __shared__ float  Osh[16 * HD];        // block accumulation
  const int wave = threadIdx.x >> 5, lane = threadIdx.x & 31;
  const int row = lane & 15, half = lane >> 4;
  const int b = blockIdx.x >> 6, q0 = (blockIdx.x & 63) << 4;
  const float sc = 0.17677669529663687f;   // 1/sqrt(32)

  // resident Q A-operands, one per head
  const bf16_t* qb = qk + ((size_t)(b * LL + q0 + row)) * HD;
  v16bf Aq[HH];
#pragma unroll
  for (int h = 0; h < HH; ++h)
    Aq[h] = ld8x2(qb + h * DHH + half * 16, qb + h * DHH + 16 + half * 8);

  v8f acc[HH][2];
#pragma unroll
  for (int h = 0; h < HH; ++h) { acc[h][0] = {}; acc[h][1] = {}; }

  for (int step = 0; step < 4; ++step) {
    const int k0 = wave * 128 + step * 32;
    if (step < 3)
      __builtin_prefetch(qk + ((size_t)(b * LL + k0 + 32)) * HD, 0, 0);

    // scores: S[h][t] = Q_h (16x32) x K_h^T (32x16)
    v8f S[HH][2];
#pragma unroll
    for (int h = 0; h < HH; ++h)
#pragma unroll
      for (int t = 0; t < 2; ++t) {
        const bf16_t* kb = qk + ((size_t)(b * LL + k0 + t * 16 + row)) * HD + h * DHH + half * 16;
        v8f z = {};
        S[h][t] = wmma_bf(Aq[h], ld16(kb), z);
      }

    // softmax over heads: all 8 head values for (q,k) live in the same lane/slot
#pragma unroll
    for (int t = 0; t < 2; ++t)
#pragma unroll
      for (int j = 0; j < 8; ++j) {
        float m = -1e30f;
#pragma unroll
        for (int h = 0; h < HH; ++h) m = fmaxf(m, S[h][t][j] * sc);
        float e[HH], sum = 0.f;
#pragma unroll
        for (int h = 0; h < HH; ++h) { e[h] = __expf(S[h][t][j] * sc - m); sum += e[h]; }
        const float rs = 1.f / sum;
#pragma unroll
        for (int h = 0; h < HH; ++h) S[h][t][j] = e[h] * rs;
      }

    // per head: D-layout -> A-layout via per-wave LDS, then P @ V
#pragma unroll
    for (int h = 0; h < HH; ++h) {
      bf16_t* pl = p_lds[wave];
#pragma unroll
      for (int t = 0; t < 2; ++t)
#pragma unroll
        for (int j = 0; j < 8; ++j)
          pl[(j + 8 * half) * 40 + t * 16 + row] = (bf16_t)S[h][t][j];
      // same-wave LDS ordering: compiler inserts s_wait_dscnt on the aliasing loads
      v16bf Ap = ld8x2(pl + row * 40 + half * 16, pl + row * 40 + 16 + half * 8);
#pragma unroll
      for (int t2 = 0; t2 < 2; ++t2) {
        const bf16_t* vb = vT + ((size_t)(b * HD + h * DHH + t2 * 16 + row)) * LL + k0 + half * 16;
        acc[h][t2] = wmma_bf(Ap, ld16(vb), acc[h][t2]);
      }
    }
  }

  // deterministic sequential reduction over the 8 k-split waves
  for (int w = 0; w < 8; ++w) {
    if (wave == w) {
#pragma unroll
      for (int h = 0; h < HH; ++h)
#pragma unroll
        for (int t2 = 0; t2 < 2; ++t2)
#pragma unroll
          for (int j = 0; j < 8; ++j) {
            const int idx = (j + 8 * half) * HD + h * DHH + t2 * 16 + row;
            if (w == 0) Osh[idx] = acc[h][t2][j];
            else        Osh[idx] += acc[h][t2][j];
          }
    }
    __syncthreads();
  }
  for (int e2 = threadIdx.x; e2 < 16 * HD; e2 += 256)
    out_pre[((size_t)(b * LL + q0 + (e2 >> 8))) * HD + (e2 & 255)] = Osh[e2];
}

// ---------------- kernel 5: sparse-bias correction (exact last-write-wins) ----------------
// one block per (b,q): out_pre[q] += sum_{distinct biased k} (P_biased - P_unbiased) * V[k]
__global__ __launch_bounds__(256) void k_corr(
    const int* __restrict__ E_idx, const float* __restrict__ bproj,
    const bf16_t* __restrict__ qk, const bf16_t* __restrict__ val,
    float* __restrict__ out_pre) {
  __shared__ float s0[II * HH];
  __shared__ float bi[II * HH];
  __shared__ float dl[II * HH];
  __shared__ int   ki[II];
  __shared__ int   fl[II];
  const int b = blockIdx.x >> 10, q = blockIdx.x & (LL - 1);
  const int t = threadIdx.x, i = t >> 3, h = t & 7;
  const size_t rbase = (size_t)(b * LL + q);
  const float sc = 0.17677669529663687f;

  const int k = E_idx[rbase * II + i];
  const bf16_t* qr = qk + rbase * HD + h * DHH;
  const bf16_t* kr = qk + ((size_t)(b * LL + k)) * HD + h * DHH;
  float s = 0.f;
#pragma unroll
  for (int d = 0; d < DHH; ++d) s += (float)qr[d] * (float)kr[d];
  s0[t] = s * sc;
  bi[t] = bproj[(rbase * II + i) * HH + h];
  if (h == 0) ki[i] = k;
  __syncthreads();

  if (t < II) {
    const int myk = ki[t];
    int last = 1;
    for (int i2 = t + 1; i2 < II; ++i2) if (ki[i2] == myk) last = 0;
    fl[t] = last;
    float a0[HH], a1[HH], m0 = -1e30f, m1 = -1e30f;
#pragma unroll
    for (int hh = 0; hh < HH; ++hh) {
      a0[hh] = s0[t * HH + hh];
      a1[hh] = a0[hh] + bi[t * HH + hh];
      m0 = fmaxf(m0, a0[hh]); m1 = fmaxf(m1, a1[hh]);
    }
    float e0[HH], e1[HH], z0 = 0.f, z1 = 0.f;
#pragma unroll
    for (int hh = 0; hh < HH; ++hh) {
      e0[hh] = __expf(a0[hh] - m0); z0 += e0[hh];
      e1[hh] = __expf(a1[hh] - m1); z1 += e1[hh];
    }
    const float r0 = 1.f / z0, r1 = 1.f / z1;
#pragma unroll
    for (int hh = 0; hh < HH; ++hh) dl[t * HH + hh] = e1[hh] * r1 - e0[hh] * r0;
  }
  __syncthreads();

  const int hh = t >> 5;   // thread t = output feature, its head
  float a = 0.f;
  for (int i2 = 0; i2 < II; ++i2)
    if (fl[i2]) a += dl[i2 * HH + hh] * (float)val[((size_t)(b * LL + ki[i2])) * HD + t];
  out_pre[rbase * HD + t] += a;
}

// ---------------- kernel 6: gate * out_pre, then @ Wo^T + bo ----------------
__global__ __launch_bounds__(32) void k_final(
    const float* __restrict__ out_pre, const float* __restrict__ gate,
    const bf16_t* __restrict__ wo, const float* __restrict__ bo,
    float* __restrict__ out) {
  __shared__ bf16_t a_lds[16 * 264];
  const int lane = threadIdx.x, row = lane & 15, half = lane >> 4;
  const int b = blockIdx.x >> 6, q0 = (blockIdx.x & 63) << 4;
  const size_t base = ((size_t)(b * LL + q0 + row)) * HD + half * 128;
#pragma unroll
  for (int c4 = 0; c4 < 32; ++c4) {
    const float4 o = ((const float4*)(out_pre + base))[c4];
    const float4 g = ((const float4*)(gate + base))[c4];
    bf16_t* dst = &a_lds[row * 264 + half * 128 + c4 * 4];
    dst[0] = (bf16_t)(o.x * g.x); dst[1] = (bf16_t)(o.y * g.y);
    dst[2] = (bf16_t)(o.z * g.z); dst[3] = (bf16_t)(o.w * g.w);
  }
  __syncthreads();
  v16bf A[8];
#pragma unroll
  for (int ks = 0; ks < 8; ++ks) {
    const bf16_t* p = &a_lds[row * 264 + ks * 32 + half * 16];
    A[ks] = ld8x2(p, p + 16);
  }
#pragma unroll
  for (int nt = 0; nt < 8; ++nt) {
    v8f acc = {};
#pragma unroll
    for (int ks = 0; ks < 8; ++ks)
      acc = wmma_bf(A[ks], ld16(wo + (size_t)(nt * 16 + row) * HD + ks * 32 + half * 16), acc);
    const int f = nt * 16 + row;
    const float bov = bo[f];
#pragma unroll
    for (int j = 0; j < 8; ++j)
      out[((size_t)(b * LL + q0 + j + 8 * half)) * DD + f] = acc[j] + bov;
  }
}

// ---------------- launch ----------------
extern "C" void kernel_launch(void* const* d_in, const int* in_sizes, int n_in,
                              void* d_out, int out_size, void* d_ws, size_t ws_size,
                              hipStream_t stream) {
  const float* x    = (const float*)d_in[0];
  const float* bias = (const float*)d_in[1];
  const int*   E    = (const int*)d_in[2];
  // d_in[3] mask_attend: unused by reference math
  const float* g_in = (const float*)d_in[4];
  const float* b_in = (const float*)d_in[5];
  const float* g_b  = (const float*)d_in[6];
  const float* b_b  = (const float*)d_in[7];
  const float* Wqk  = (const float*)d_in[8];
  const float* Wv   = (const float*)d_in[9];
  const float* Wb   = (const float*)d_in[10];
  const float* Wg   = (const float*)d_in[11];
  const float* bg   = (const float*)d_in[12];
  const float* Wo   = (const float*)d_in[13];
  const float* bo   = (const float*)d_in[14];
  float* out = (float*)d_out;

  char* ws = (char*)d_ws;
  const size_t WN = (size_t)HD * DD * sizeof(bf16_t);   // 65536 per weight
  bf16_t* wqk_b = (bf16_t*)(ws);
  bf16_t* wv_b  = (bf16_t*)(ws + WN);
  bf16_t* wg_b  = (bf16_t*)(ws + 2 * WN);
  bf16_t* wo_b  = (bf16_t*)(ws + 3 * WN);
  size_t off = 4 * WN;
  bf16_t* qk_b  = (bf16_t*)(ws + off); off += (size_t)BB * LL * HD * sizeof(bf16_t);
  bf16_t* val_b = (bf16_t*)(ws + off); off += (size_t)BB * LL * HD * sizeof(bf16_t);
  bf16_t* vT_b  = (bf16_t*)(ws + off); off += (size_t)BB * LL * HD * sizeof(bf16_t);
  float*  gate  = (float*)(ws + off);  off += (size_t)BB * LL * HD * sizeof(float);
  float*  bproj = (float*)(ws + off);  off += (size_t)BB * LL * II * HH * sizeof(float);
  float*  opre  = (float*)(ws + off);  off += (size_t)BB * LL * HD * sizeof(float);
  (void)ws_size; (void)in_sizes; (void)n_in; (void)out_size;

  k_wcvt <<<128, 256, 0, stream>>>(Wqk, Wv, Wg, Wo, wqk_b, wv_b, wg_b, wo_b);
  k_tok  <<<BB * (LL / 16), 32, 0, stream>>>(x, g_in, b_in, wqk_b, wv_b, wg_b, bg,
                                             qk_b, val_b, vT_b, gate);
  k_bias <<<256, 256, 0, stream>>>(bias, g_b, b_b, Wb, bproj);
  k_attn <<<BB * (LL / 16), 256, 0, stream>>>(qk_b, vT_b, opre);
  k_corr <<<BB * LL, 256, 0, stream>>>(E, bproj, qk_b, val_b, opre);
  k_final<<<BB * (LL / 16), 32, 0, stream>>>(opre, gate, wo_b, bo, out);
}
